// MambaBlock_6021544149641
// MI455X (gfx1250) — matrix operand order repaired and checked by reading, not compiled, not run on hardware
//
#include <hip/hip_runtime.h>
#include <cstdint>
#include <cstddef>

// ---------------------------------------------------------------------------
// Problem constants (fixed by the reference)
// ---------------------------------------------------------------------------
#define BB      2
#define LL      2048
#define DMODEL  1024
#define DINNER  2048
#define DCONV   4
#define DTRANK  64
#define DSTATE  16
#define MROWS   (BB * LL)          // 4096
#define NXZ     (2 * DINNER)       // 4096
#define NXDBL_P 128                // dt_rank + 2*d_state = 96, padded to 128

typedef __attribute__((ext_vector_type(16))) __bf16 v16bf;
typedef __attribute__((ext_vector_type(8)))  __bf16 v8bf;
typedef __attribute__((ext_vector_type(8)))  float  v8f;

// ---------------------------------------------------------------------------
// bf16 helpers (bit-exact RNE, no reliance on __bf16 arithmetic lowering)
// ---------------------------------------------------------------------------
__device__ __forceinline__ __bf16 f2bf(float f) {
    union { float f; unsigned u; } cv; cv.f = f;
    unsigned r = (cv.u + 0x7FFFu + ((cv.u >> 16) & 1u)) >> 16;
    union { unsigned short s; __bf16 b; } o; o.s = (unsigned short)r;
    return o.b;
}
__device__ __forceinline__ float bf2f(__bf16 b) {
    union { unsigned short s; __bf16 b; } i; i.b = b;
    union { unsigned u; float f; } cv; cv.u = ((unsigned)i.s) << 16;
    return cv.f;
}
__device__ __forceinline__ float sigmoidf_fast(float x) {
    return 1.0f / (1.0f + __expf(-x));
}

// ---------------------------------------------------------------------------
// Elementwise f32 -> bf16 cast
// ---------------------------------------------------------------------------
__global__ void __launch_bounds__(256)
cast_f32_bf16_kernel(const float* __restrict__ in, __bf16* __restrict__ out, int n) {
    int id = blockIdx.x * 256 + threadIdx.x;
    if (id < n) out[id] = f2bf(in[id]);
}

// ---------------------------------------------------------------------------
// W [K x N] f32 (row-major)  ->  Wt [Npad x K] bf16 (row-major, K contiguous)
// Pad rows (n >= N) are zero-filled so padded GEMM columns produce zeros.
// ---------------------------------------------------------------------------
__global__ void __launch_bounds__(256)
transpose_cast_kernel(const float* __restrict__ W, __bf16* __restrict__ Wt,
                      int K, int N, int Npad) {
    int id = blockIdx.x * 256 + threadIdx.x;
    int total = K * Npad;
    if (id >= total) return;
    int k = id / Npad;
    int n = id - k * Npad;
    float v = (n < N) ? W[(size_t)k * N + n] : 0.0f;
    Wt[(size_t)n * K + k] = f2bf(v);
}

// ---------------------------------------------------------------------------
// Generic bf16 WMMA GEMM:  C[M x N] f32 = A[M x K] bf16 @ Bt[N x K]^T bf16
//   block = 256 threads = 8 waves; block tile 64(M) x 128(N)
//   wave tile 32x32: 2 A-frags x 2 B-frags -> 4 v_wmma_f32_16x16x32_bf16 / k-step
//   grid = dim3(N/128, M/64); K % 32 == 0; lda/ldb mult of 16 (alignment)
// A fragment (16-bit 16x32): lane m = lane&15, K-half = lane>>4
//   elems 0..7  = K[kh*8   .. kh*8+7]   (16B load)
//   elems 8..15 = K[16+kh*8.. 16+kh*8+7](16B load)
// B fragment (16-bit 32x16): lane n = lane&15, same K order as A (two 16B loads)
// ---------------------------------------------------------------------------
__global__ void __launch_bounds__(256)
gemm_bf16_wmma_kernel(const __bf16* __restrict__ A, const __bf16* __restrict__ Bt,
                      float* __restrict__ C, int K, int lda, int ldb, int ldc) {
    const int lane = threadIdx.x & 31;
    const int wave = threadIdx.x >> 5;
    const int wm   = wave & 1;                       // 2 wave rows
    const int wn   = wave >> 1;                      // 4 wave cols
    const int m0   = blockIdx.y * 64  + wm * 32;
    const int n0   = blockIdx.x * 128 + wn * 32;
    const int lm   = lane & 15;                      // row / col within 16-tile
    const int kh   = lane >> 4;                      // K-half select

    v8f acc[2][2] = {};

    const __bf16* Ar0 = A  + (size_t)(m0 +      lm) * lda + kh * 8;
    const __bf16* Ar1 = A  + (size_t)(m0 + 16 + lm) * lda + kh * 8;
    const __bf16* Br0 = Bt + (size_t)(n0 +      lm) * ldb + kh * 8;
    const __bf16* Br1 = Bt + (size_t)(n0 + 16 + lm) * ldb + kh * 8;

    for (int k0 = 0; k0 < K; k0 += 32) {
        v8bf a0lo = *(const v8bf*)(Ar0 + k0);
        v8bf a0hi = *(const v8bf*)(Ar0 + k0 + 16);
        v8bf a1lo = *(const v8bf*)(Ar1 + k0);
        v8bf a1hi = *(const v8bf*)(Ar1 + k0 + 16);
        v16bf a0 = __builtin_shufflevector(a0lo, a0hi,
                     0,1,2,3,4,5,6,7,8,9,10,11,12,13,14,15);
        v16bf a1 = __builtin_shufflevector(a1lo, a1hi,
                     0,1,2,3,4,5,6,7,8,9,10,11,12,13,14,15);
        v16bf b0 = __builtin_shufflevector(*(const v8bf*)(Br0 + k0), *(const v8bf*)(Br0 + k0 + 16),
                     0,1,2,3,4,5,6,7,8,9,10,11,12,13,14,15);
        v16bf b1 = __builtin_shufflevector(*(const v8bf*)(Br1 + k0), *(const v8bf*)(Br1 + k0 + 16),
                     0,1,2,3,4,5,6,7,8,9,10,11,12,13,14,15);

        acc[0][0] = __builtin_amdgcn_wmma_f32_16x16x32_bf16(
                        false, a0, false, b0, (short)0, acc[0][0], false, false);
        acc[0][1] = __builtin_amdgcn_wmma_f32_16x16x32_bf16(
                        false, a0, false, b1, (short)0, acc[0][1], false, false);
        acc[1][0] = __builtin_amdgcn_wmma_f32_16x16x32_bf16(
                        false, a1, false, b0, (short)0, acc[1][0], false, false);
        acc[1][1] = __builtin_amdgcn_wmma_f32_16x16x32_bf16(
                        false, a1, false, b1, (short)0, acc[1][1], false, false);
    }

    // C/D layout: VGPR j <-> row (lane>>4)*8 + j, col lane&15
    const int rbase = kh * 8;
    const int col   = lm;
#pragma unroll
    for (int r = 0; r < 2; ++r)
#pragma unroll
        for (int c = 0; c < 2; ++c)
#pragma unroll
            for (int j = 0; j < 8; ++j)
                C[(size_t)(m0 + 16 * r + rbase + j) * ldc + (n0 + 16 * c + col)]
                    = acc[r][c][j];
}

// ---------------------------------------------------------------------------
// Causal depthwise conv1d (kernel 4) + bias + SiLU over xb = xz[:, :, 0:2048]
// Writes bf16 activations [MROWS x DINNER] for the next GEMM / scan.
// ---------------------------------------------------------------------------
__global__ void __launch_bounds__(256)
conv_silu_kernel(const float* __restrict__ xz, const float* __restrict__ cw,
                 const float* __restrict__ cb, __bf16* __restrict__ xb) {
    int id = blockIdx.x * 256 + threadIdx.x;          // < MROWS*DINNER
    if (id >= MROWS * DINNER) return;
    int d = id & (DINNER - 1);
    int l = (id >> 11) & (LL - 1);
    int b = id >> 22;
    float acc = cb[d];
#pragma unroll
    for (int k = 0; k < DCONV; ++k) {
        int ll = l - (DCONV - 1) + k;
        if (ll >= 0)
            acc += xz[(((size_t)b * LL + ll) * NXZ) + d] * cw[d * DCONV + k];
    }
    xb[id] = f2bf(acc * sigmoidf_fast(acc));
}

// ---------------------------------------------------------------------------
// Extract dt_low (cols 0..63 of x_dbl, ld=128) as bf16 [MROWS x 64]
// ---------------------------------------------------------------------------
__global__ void __launch_bounds__(256)
dtlow_cast_kernel(const float* __restrict__ xdbl, __bf16* __restrict__ dtlow) {
    int id = blockIdx.x * 256 + threadIdx.x;          // < MROWS*DTRANK
    if (id >= MROWS * DTRANK) return;
    int row = id >> 6;
    int c   = id & (DTRANK - 1);
    dtlow[id] = f2bf(xdbl[(size_t)row * NXDBL_P + c]);
}

// ---------------------------------------------------------------------------
// dt = softplus(dtproj + b_dt)  (in place)
// ---------------------------------------------------------------------------
__global__ void __launch_bounds__(256)
softplus_kernel(float* __restrict__ dt, const float* __restrict__ bdt) {
    int id = blockIdx.x * 256 + threadIdx.x;          // < MROWS*DINNER
    if (id >= MROWS * DINNER) return;
    int d = id & (DINNER - 1);
    float x = dt[id] + bdt[d];
    dt[id] = (x > 20.0f) ? x : __logf(1.0f + __expf(x));
}

// ---------------------------------------------------------------------------
// Selective scan: one thread per (b, d) channel, 16 states in registers.
// Fused epilogue: y = scan + xb*D;  out_pre = y * silu(z)  -> bf16
// grid = (DINNER/256, BB), block = 256
// ---------------------------------------------------------------------------
__global__ void __launch_bounds__(256)
scan_kernel(const float* __restrict__ dt, const float* __restrict__ xdbl,
            const __bf16* __restrict__ xb, const float* __restrict__ xz,
            const float* __restrict__ A_log, const float* __restrict__ Dp,
            __bf16* __restrict__ out_pre) {
    int d = blockIdx.x * 256 + threadIdx.x;           // 0..DINNER-1
    int b = blockIdx.y;                               // 0..BB-1

    float a[DSTATE];
#pragma unroll
    for (int n = 0; n < DSTATE; ++n)
        a[n] = -__expf(A_log[(size_t)d * DSTATE + n]);

    float h[DSTATE];
#pragma unroll
    for (int n = 0; n < DSTATE; ++n) h[n] = 0.0f;

    const float Dv = Dp[d];

    for (int l = 0; l < LL; ++l) {
        const size_t row = (size_t)b * LL + l;
        const float dtv = dt[row * DINNER + d];
        const float xv  = bf2f(xb[row * DINNER + d]);
        const float xdt = xv * dtv;
        const float* bc = xdbl + row * NXDBL_P + DTRANK;   // B_t[0..15], C_t[16..31]
        float y = 0.0f;
#pragma unroll
        for (int n = 0; n < DSTATE; ++n) {
            float dA = __expf(a[n] * dtv);
            h[n] = h[n] * dA + xdt * bc[n];
            y += h[n] * bc[DSTATE + n];
        }
        const float zv = xz[row * NXZ + DINNER + d];
        const float yv = y + xv * Dv;
        out_pre[row * DINNER + d] = f2bf(yv * (zv * sigmoidf_fast(zv)));
    }
}

// ---------------------------------------------------------------------------
// Host launcher
// ---------------------------------------------------------------------------
extern "C" void kernel_launch(void* const* d_in, const int* in_sizes, int n_in,
                              void* d_out, int out_size, void* d_ws, size_t ws_size,
                              hipStream_t stream) {
    (void)in_sizes; (void)n_in; (void)out_size; (void)ws_size;

    const float* x     = (const float*)d_in[0];
    const float* W_in  = (const float*)d_in[1];
    const float* convw = (const float*)d_in[2];
    const float* convb = (const float*)d_in[3];
    const float* W_x   = (const float*)d_in[4];
    const float* W_dt  = (const float*)d_in[5];
    const float* b_dt  = (const float*)d_in[6];
    const float* A_log = (const float*)d_in[7];
    const float* Dp    = (const float*)d_in[8];
    const float* W_out = (const float*)d_in[9];
    float* out = (float*)d_out;

    char* ws = (char*)d_ws;
    size_t off = 0;
    auto alloc = [&](size_t bytes) -> char* {
        off = (off + 255) & ~(size_t)255;
        char* p = ws + off;
        off += bytes;
        return p;
    };

    __bf16* x_bf    = (__bf16*)alloc((size_t)MROWS * DMODEL * 2);
    __bf16* Wt_in   = (__bf16*)alloc((size_t)NXZ * DMODEL * 2);
    __bf16* Wt_x    = (__bf16*)alloc((size_t)NXDBL_P * DINNER * 2);
    __bf16* Wt_dt   = (__bf16*)alloc((size_t)DINNER * DTRANK * 2);
    __bf16* Wt_out  = (__bf16*)alloc((size_t)DMODEL * DINNER * 2);
    float*  xz      = (float*) alloc((size_t)MROWS * NXZ * 4);
    __bf16* xb_bf   = (__bf16*)alloc((size_t)MROWS * DINNER * 2);
    float*  x_dbl   = (float*) alloc((size_t)MROWS * NXDBL_P * 4);
    __bf16* dtlo_bf = (__bf16*)alloc((size_t)MROWS * DTRANK * 2);
    float*  dtbuf   = (float*) alloc((size_t)MROWS * DINNER * 4);
    __bf16* pre_bf  = (__bf16*)alloc((size_t)MROWS * DINNER * 2);

    const dim3 blk(256);

    // Precision staging: x and weights -> bf16 (weights transposed, K-contiguous)
    cast_f32_bf16_kernel<<<(MROWS * DMODEL) / 256, blk, 0, stream>>>(
        x, x_bf, MROWS * DMODEL);
    transpose_cast_kernel<<<(DMODEL * NXZ) / 256, blk, 0, stream>>>(
        W_in, Wt_in, DMODEL, NXZ, NXZ);
    transpose_cast_kernel<<<(DINNER * NXDBL_P) / 256, blk, 0, stream>>>(
        W_x, Wt_x, DINNER, DTRANK + 2 * DSTATE, NXDBL_P);
    transpose_cast_kernel<<<(DTRANK * DINNER) / 256, blk, 0, stream>>>(
        W_dt, Wt_dt, DTRANK, DINNER, DINNER);
    transpose_cast_kernel<<<(DINNER * DMODEL) / 256, blk, 0, stream>>>(
        W_out, Wt_out, DINNER, DMODEL, DMODEL);

    // GEMM1: xz = x @ W_in            [4096 x 1024] @ [1024 x 4096]
    gemm_bf16_wmma_kernel<<<dim3(NXZ / 128, MROWS / 64), blk, 0, stream>>>(
        x_bf, Wt_in, xz, DMODEL, DMODEL, DMODEL, NXZ);

    // Conv + SiLU -> xb (bf16)
    conv_silu_kernel<<<(MROWS * DINNER) / 256, blk, 0, stream>>>(
        xz, convw, convb, xb_bf);

    // GEMM2: x_dbl = xb @ W_x         [4096 x 2048] @ [2048 x 96(pad 128)]
    gemm_bf16_wmma_kernel<<<dim3(NXDBL_P / 128, MROWS / 64), blk, 0, stream>>>(
        xb_bf, Wt_x, x_dbl, DINNER, DINNER, DINNER, NXDBL_P);

    // dt_low -> bf16
    dtlow_cast_kernel<<<(MROWS * DTRANK) / 256, blk, 0, stream>>>(x_dbl, dtlo_bf);

    // GEMM3: dtproj = dt_low @ W_dt   [4096 x 64] @ [64 x 2048]
    gemm_bf16_wmma_kernel<<<dim3(DINNER / 128, MROWS / 64), blk, 0, stream>>>(
        dtlo_bf, Wt_dt, dtbuf, DTRANK, DTRANK, DTRANK, DINNER);

    // dt = softplus(dtproj + b_dt)
    softplus_kernel<<<(MROWS * DINNER) / 256, blk, 0, stream>>>(dtbuf, b_dt);

    // Selective scan + fused (y + xb*D) * silu(z) epilogue -> bf16
    scan_kernel<<<dim3(DINNER / 256, BB), blk, 0, stream>>>(
        dtbuf, x_dbl, xb_bf, xz, A_log, Dp, pre_bf);

    // GEMM4: out = pre @ W_out        [4096 x 2048] @ [2048 x 1024] -> f32 d_out
    gemm_bf16_wmma_kernel<<<dim3(DMODEL / 128, MROWS / 64), blk, 0, stream>>>(
        pre_bf, Wt_out, out, DINNER, DINNER, DINNER, DMODEL);
}
